// SNN_54666343744030
// MI455X (gfx1250) — matrix operand order repaired and checked by reading, not compiled, not run on hardware
//
#include <hip/hip_runtime.h>
#include <hip/hip_bf16.h>

// ---------------------------------------------------------------------------
// Fused SNN (Leaky LIF, 2-layer MLP) scan kernel for gfx1250 (MI455X).
//   layer1: cur1 = x_t @ W1^T + b1   -> bf16 WMMA 16x16x32, f32 accum
//   LIF1 in registers (WMMA C/D layout), spikes (exact 0/1) -> LDS bf16
//   layer2: cur2 = spk1 @ W2^T + b2  -> bf16 WMMA on wave 0 (N padded 5->16)
//   LIF2 + spike-count accumulation in wave0 registers.
// One block = 16 batch rows, scans all T=512 steps; grid = 64 blocks.
// ---------------------------------------------------------------------------

typedef __attribute__((ext_vector_type(16))) __bf16 v16bf;
typedef __attribute__((ext_vector_type(8)))  __bf16 v8bf;
typedef __attribute__((ext_vector_type(8)))  float  v8f;
typedef __attribute__((ext_vector_type(4)))  float  v4f;

#define BATCH   1024
#define TSTEPS  512
#define DIN     128
#define HDIM    256
#define DOUT    5
#define BETA    0.9f
#define THR     1.0f

#define MTILE   16
#define XPITCH  136   // 128 + 8 bf16 (16B) pad: rotates LDS banks per row
#define SPITCH  264   // 256 + 8 bf16 pad

// A-matrix fragment (16x32 bf16) from LDS.
// Layout: lanes 0-15 hold M=lane with K=klo..klo+15, klo=0;
//         lanes 16-31 hold M=lane-16 with klo=16.  p points at element klo.
__device__ __forceinline__ v16bf a_frag_lds(const __bf16* p) {
  v8bf lo = *(const v8bf*)p;
  v8bf hi = *(const v8bf*)(p + 8);
  v16bf r;
#pragma unroll
  for (int i = 0; i < 8; ++i) { r[i] = lo[i]; r[i + 8] = hi[i]; }
  return r;
}

// B-matrix fragment (32x16 bf16) from 16 consecutive f32 (row-major weights).
// Contiguous K (same as A): lanes 0-15 hold N=lane, K=0..15; lanes 16-31 hold N=lane-16,
// K=16..31 -> per lane 16 consecutive K values starting at kb.
__device__ __forceinline__ v16bf b_frag_f32(const float* p) {
  v16bf r;
#pragma unroll
  for (int i = 0; i < 16; ++i) r[i] = (__bf16)p[i];
  return r;
}

__global__ __launch_bounds__(256) void
snn_leaky_fused(const float* __restrict__ x,
                const float* __restrict__ W1, const float* __restrict__ b1,
                const float* __restrict__ W2, const float* __restrict__ b2,
                float* __restrict__ out)
{
  __shared__ __bf16 xb[2][MTILE * XPITCH];   // x tile, bf16, double-buffered
  __shared__ __bf16 sb[2][MTILE * SPITCH];   // layer-1 spikes, double-buffered

  const int tid  = threadIdx.x;
  const int wave = tid >> 5;
  const int lane = tid & 31;
  const int lm   = lane & 15;
  const int ka   = (lane < 16) ? 0 : 16;   // A-fragment K base
  const int kb   = (lane < 16) ? 0 : 16;   // B-fragment K base
  const int rowb = (lane < 16) ? 0 : 8;    // C/D row base
  const int b0   = blockIdx.x * MTILE;

  // ---- W1 B-fragments: converted once, VGPR-resident for all 512 steps ----
  v16bf w1f[2][4];
#pragma unroll
  for (int n = 0; n < 2; ++n) {
    const float* wr = W1 + (size_t)(wave * 32 + n * 16 + lm) * DIN;
#pragma unroll
    for (int c = 0; c < 4; ++c)
      w1f[n][c] = b_frag_f32(wr + c * 32 + kb);
  }
  const float bias1_0 = b1[wave * 32 + lm];
  const float bias1_1 = b1[wave * 32 + 16 + lm];

  // ---- wave0 only: W2 B-fragments (8 K-chunks), columns >= DOUT are zero ----
  v16bf w2f[8];
  float bias2 = 0.0f;
  if (wave == 0) {
#pragma unroll
    for (int c = 0; c < 8; ++c) {
      v16bf f;
#pragma unroll
      for (int i = 0; i < 16; ++i) f[i] = (__bf16)0.0f;
      if (lm < DOUT)
        f = b_frag_f32(W2 + (size_t)lm * HDIM + c * 32 + kb);
      w2f[c] = f;
    }
    if (lm < DOUT) bias2 = b2[lm];
  }

  // ---- persistent recurrent state (WMMA C/D register layout) ----
  v8f mem1a = {}, mem1b = {};      // layer-1 membrane, two 16x16 tiles
  v8f mem2  = {}, acc_o = {};      // wave0: layer-2 membrane + spike counts

  const int srow = tid >> 4;       // staging: row 0..15
  const int sseg = tid & 15;       // staging: 8-float segment
  const float* xbase = x + ((size_t)(b0 + srow) * TSTEPS) * DIN + sseg * 8;

  for (int t = 0; t < TSTEPS; ++t) {
    const int pb = t & 1;

    // ---- stage x[:, t, :] (16x128 f32) into LDS as bf16, coalesced ----
    {
      const float* p = xbase + (size_t)t * DIN;
      v4f a = *(const v4f*)p;
      v4f b = *(const v4f*)(p + 4);
      v8bf v;
      v[0] = (__bf16)a[0]; v[1] = (__bf16)a[1];
      v[2] = (__bf16)a[2]; v[3] = (__bf16)a[3];
      v[4] = (__bf16)b[0]; v[5] = (__bf16)b[1];
      v[6] = (__bf16)b[2]; v[7] = (__bf16)b[3];
      *(v8bf*)&xb[pb][srow * XPITCH + sseg * 8] = v;
    }
    __syncthreads();

    // ---- layer 1: each wave computes 16 rows x 32 cols of cur1 ----
    v8f acc0 = {}, acc1 = {};
#pragma unroll
    for (int c = 0; c < 4; ++c) {
      v16bf af = a_frag_lds(&xb[pb][lm * XPITCH + c * 32 + ka]);
      acc0 = __builtin_amdgcn_wmma_f32_16x16x32_bf16(false, af, false, w1f[0][c],
                                                     (short)0, acc0, false, false);
      acc1 = __builtin_amdgcn_wmma_f32_16x16x32_bf16(false, af, false, w1f[1][c],
                                                     (short)0, acc1, false, false);
    }

    // ---- LIF layer 1 (subtract reset) + spike store (exact 0/1 bf16) ----
    __bf16* sdst = &sb[pb][rowb * SPITCH + wave * 32 + lm];
#pragma unroll
    for (int r = 0; r < 8; ++r) {
      float m0  = mem1a[r];
      float rs0 = (m0 > THR) ? THR : 0.0f;
      m0 = BETA * m0 + (acc0[r] + bias1_0) - rs0;
      mem1a[r] = m0;
      sdst[r * SPITCH] = (m0 > THR) ? (__bf16)1.0f : (__bf16)0.0f;

      float m1  = mem1b[r];
      float rs1 = (m1 > THR) ? THR : 0.0f;
      m1 = BETA * m1 + (acc1[r] + bias1_1) - rs1;
      mem1b[r] = m1;
      sdst[r * SPITCH + 16] = (m1 > THR) ? (__bf16)1.0f : (__bf16)0.0f;
    }
    __syncthreads();

    // ---- layer 2 (wave 0): spk1 @ W2^T, LIF2, spike-count accumulate ----
    if (wave == 0) {
      v8f acc2 = {};
#pragma unroll
      for (int c = 0; c < 8; ++c) {
        v16bf af = a_frag_lds(&sb[pb][lm * SPITCH + c * 32 + ka]);
        acc2 = __builtin_amdgcn_wmma_f32_16x16x32_bf16(false, af, false, w2f[c],
                                                       (short)0, acc2, false, false);
      }
#pragma unroll
      for (int r = 0; r < 8; ++r) {
        float m  = mem2[r];
        float rs = (m > THR) ? THR : 0.0f;
        m = BETA * m + (acc2[r] + bias2) - rs;
        mem2[r] = m;
        acc_o[r] += (m > THR) ? 1.0f : 0.0f;
      }
    }
  }

  // ---- write spike counts: [BATCH, DOUT] f32 ----
  if (wave == 0 && lm < DOUT) {
#pragma unroll
    for (int r = 0; r < 8; ++r)
      out[(size_t)(b0 + rowb + r) * DOUT + lm] = acc_o[r];
  }
}

extern "C" void kernel_launch(void* const* d_in, const int* in_sizes, int n_in,
                              void* d_out, int out_size, void* d_ws, size_t ws_size,
                              hipStream_t stream) {
  (void)in_sizes; (void)n_in; (void)out_size; (void)d_ws; (void)ws_size;
  const float* x  = (const float*)d_in[0];
  const float* W1 = (const float*)d_in[1];
  const float* b1 = (const float*)d_in[2];
  const float* W2 = (const float*)d_in[3];
  const float* b2 = (const float*)d_in[4];
  float* out = (float*)d_out;

  dim3 grid(BATCH / MTILE);   // 64 blocks x 16 batch rows
  dim3 block(256);            // 8 waves (wave32)
  snn_leaky_fused<<<grid, block, 0, stream>>>(x, W1, b1, W2, b2, out);
}
